// MultiHeadAttention_48773648614069
// MI455X (gfx1250) — matrix operand order repaired and checked by reading, not compiled, not run on hardware
//
#include <hip/hip_runtime.h>
#include <stdint.h>

#define B_ 8
#define N_ 1024
#define C_ 512
#define H_ 8
#define D_ 64

typedef __attribute__((ext_vector_type(16))) __bf16 v16bf;
typedef __attribute__((ext_vector_type(8)))  float  v8f;
typedef __attribute__((ext_vector_type(4)))  int    v4i;

#if defined(__has_builtin)
#  if __has_builtin(__builtin_amdgcn_global_load_async_to_lds_b128) && \
      __has_builtin(__builtin_amdgcn_s_wait_asynccnt)
#    define HAVE_ASYNC_LDS 1
#  endif
#endif

typedef __attribute__((address_space(1))) v4i* glb_v4i_p;
typedef __attribute__((address_space(3))) v4i* lds_v4i_p;

union Frag { v16bf v; unsigned int w[8]; };

__device__ __forceinline__ unsigned short f2bf(float f) {
  union { float f; unsigned int u; } c; c.f = f;
  unsigned int r = c.u + 0x7FFFu + ((c.u >> 16) & 1u);
  return (unsigned short)(r >> 16);
}

// A fragment (16x32, 16-bit): lane L -> row M=L&15; K = {Kb..Kb+7, Kb+16..Kb+23}, Kb=(L>>4)*8
__device__ __forceinline__ void load_frag_a(Frag& f, const unsigned short* p, int Kb) {
#pragma unroll
  for (int i = 0; i < 4; ++i) {
    f.w[i]     = *(const unsigned int*)(p + Kb + 2 * i);
    f.w[4 + i] = *(const unsigned int*)(p + Kb + 16 + 2 * i);
  }
}

// B fragment (32x16, 16-bit): lane L -> col N=L&15; K = {Kb..Kb+7, Kb+16..Kb+23}, Kb=(L>>4)*8
__device__ __forceinline__ void load_frag_b(Frag& f, const unsigned short* p, int hi) {
  const unsigned int* q = (const unsigned int*)(p + hi * 8);
#pragma unroll
  for (int i = 0; i < 4; ++i) {
    f.w[i]     = q[i];
    f.w[4 + i] = q[8 + i];
  }
}

__device__ __forceinline__ v8f wmma_bf16(const Frag& a, const Frag& b, v8f c) {
  return __builtin_amdgcn_wmma_f32_16x16x32_bf16(false, a.v, false, b.v,
                                                 (short)0, c, false, false);
}

// ---------------- Kernel 0: weights fp32 -> bf16 ----------------
__global__ __launch_bounds__(256) void convert_w_kernel(
    const float* __restrict__ Wq, const float* __restrict__ Wk,
    const float* __restrict__ Wv, unsigned short* __restrict__ Wbf) {
  int i = blockIdx.x * 256 + threadIdx.x;
  if (i < C_ * C_) {
    Wbf[i]               = f2bf(Wq[i]);
    Wbf[C_ * C_ + i]     = f2bf(Wk[i]);
    Wbf[2 * C_ * C_ + i] = f2bf(Wv[i]);
  }
}

// ---------------- Kernel 1: QKV projection ----------------
// Q,K stored [B,H,N,D] bf16; V stored transposed [B,H,D,N] bf16.
__global__ __launch_bounds__(128) void qkv_kernel(
    const float* __restrict__ x, const unsigned short* __restrict__ Wbf,
    const float* __restrict__ bq, const float* __restrict__ bk,
    const float* __restrict__ bv,
    unsigned short* __restrict__ Qbf, unsigned short* __restrict__ Kbf,
    unsigned short* __restrict__ VbfT) {
  __shared__ unsigned short Xs[16][C_];
  const int rowtile = blockIdx.x;  // B*N/16 = 512 tiles of 16 rows
  const int tid = threadIdx.x;
  const int wave = tid >> 5, lane = tid & 31;
  const int M = lane & 15, hi = lane >> 4, Kb = hi * 8;

  const float* xrow = x + (size_t)rowtile * 16 * C_;
  for (int i = tid; i < 16 * C_; i += 128) Xs[i >> 9][i & 511] = f2bf(xrow[i]);
  __syncthreads();

  const int grow = rowtile * 16;
  const int b = grow >> 10;
  const int nrow0 = grow & 1023;

  for (int t = wave; t < 96; t += 4) {  // 3 projections x 32 col-tiles
    const int p = t >> 5;
    const int ct = t & 31;
    v8f acc = {};
    for (int kk = 0; kk < 16; ++kk) {
      Frag a, bf_;
      load_frag_a(a, &Xs[M][kk * 32], Kb);
      load_frag_b(bf_, Wbf + (size_t)p * C_ * C_ + (size_t)(ct * 16 + M) * C_ + kk * 32, hi);
      acc = wmma_bf16(a, bf_, acc);
    }
    const int o = ct * 16 + M;  // output column for this lane
    const float* biasp = (p == 0) ? bq : ((p == 1) ? bk : bv);
    const float bvv = biasp[o];
    const int h = o >> 6, dd = o & 63;
#pragma unroll
    for (int r = 0; r < 8; ++r) {
      const unsigned short ob = f2bf(acc[r] + bvv);
      const int n = nrow0 + r + 8 * hi;
      if (p == 0)
        Qbf[(((size_t)b * H_ + h) * N_ + n) * D_ + dd] = ob;
      else if (p == 1)
        Kbf[(((size_t)b * H_ + h) * N_ + n) * D_ + dd] = ob;
      else
        VbfT[(((size_t)b * H_ + h) * D_ + dd) * N_ + n] = ob;
    }
  }
}

// ---------------- Kernel 2: flash attention with Graphormer bias ----------------
__global__ __launch_bounds__(128) void attn_kernel(
    const unsigned short* __restrict__ Qbf, const unsigned short* __restrict__ Kbf,
    const unsigned short* __restrict__ VbfT, const float* __restrict__ bias,
    const int* __restrict__ mask, unsigned short* __restrict__ Mbf) {
  __shared__ unsigned short Ps[4][16][32];
  const int blk = blockIdx.x;  // 64 (b,h) * 16 q-blocks
  const int bh = blk >> 4, qblk = blk & 15;
  const int b = bh >> 3, h = bh & 7;
  const int wave = threadIdx.x >> 5, lane = threadIdx.x & 31;
  const int qbase = qblk * 64 + wave * 16;
  const int M = lane & 15, hi = lane >> 4, Kb = hi * 8;

  // persistent Q fragments (d = 0..31, 32..63)
  Frag qa[2];
  const unsigned short* qrow = Qbf + (((size_t)b * H_ + h) * N_ + qbase + M) * D_;
#pragma unroll
  for (int ks = 0; ks < 2; ++ks) load_frag_a(qa[ks], qrow + ks * 32, Kb);

  float mrun[8], lrun[8], alpha[8];
  v8f O[4] = {};
#pragma unroll
  for (int r = 0; r < 8; ++r) { mrun[r] = -3.0e38f; lrun[r] = 0.f; }

  const float* biasbh = bias + (size_t)bh * N_ * N_;
  const int* maskb = mask + (size_t)b * N_ * N_;
  const unsigned short* krow0 = Kbf + (((size_t)b * H_ + h) * N_) * D_;
  const unsigned short* vrow0 = VbfT + (((size_t)b * H_ + h) * D_) * N_;

  for (int kt = 0; kt < 32; ++kt) {  // 32 keys per iteration
    const int kbase = kt * 32;
    // prefetch next bias chunk into cache hierarchy
    __builtin_prefetch(biasbh + (size_t)(qbase + 8 * hi) * N_ + kbase + 32 + M, 0, 1);

    v8f S0 = {}, S1 = {};
#pragma unroll
    for (int ks = 0; ks < 2; ++ks) {
      Frag kf;
      load_frag_b(kf, krow0 + (size_t)(kbase + M) * D_ + ks * 32, hi);
      S0 = wmma_bf16(qa[ks], kf, S0);
      load_frag_b(kf, krow0 + (size_t)(kbase + 16 + M) * D_ + ks * 32, hi);
      S1 = wmma_bf16(qa[ks], kf, S1);
    }
    // bias + scale + mask; online softmax (rows live across 16-lane halves)
#pragma unroll
    for (int r = 0; r < 8; ++r) {
      const int qr = qbase + r + 8 * hi;
      const int c0 = kbase + M, c1 = kbase + 16 + M;
      float t0 = (S0[r] + biasbh[(size_t)qr * N_ + c0]) * 0.125f;
      float t1 = (S1[r] + biasbh[(size_t)qr * N_ + c1]) * 0.125f;
      t0 = (maskb[(size_t)qr * N_ + c0] == 0) ? -1.0e9f : t0;
      t1 = (maskb[(size_t)qr * N_ + c1] == 0) ? -1.0e9f : t1;
      float rm = fmaxf(t0, t1);
#pragma unroll
      for (int d = 8; d >= 1; d >>= 1) rm = fmaxf(rm, __shfl_xor(rm, d, 32));
      const float mnew = fmaxf(mrun[r], rm);
      alpha[r] = __expf(mrun[r] - mnew);
      mrun[r] = mnew;
      const float p0 = __expf(t0 - mnew);
      const float p1 = __expf(t1 - mnew);
      float rs = p0 + p1;
#pragma unroll
      for (int d = 8; d >= 1; d >>= 1) rs += __shfl_xor(rs, d, 32);
      lrun[r] = lrun[r] * alpha[r] + rs;
      Ps[wave][r + 8 * hi][M] = f2bf(p0);
      Ps[wave][r + 8 * hi][16 + M] = f2bf(p1);
    }
#pragma unroll
    for (int t = 0; t < 4; ++t)
#pragma unroll
      for (int r = 0; r < 8; ++r) O[t][r] *= alpha[r];

    asm volatile("s_wait_dscnt 0" ::: "memory");  // wave-local LDS RAW
    Frag pa;
    load_frag_a(pa, &Ps[wave][M][0], Kb);
#pragma unroll
    for (int t = 0; t < 4; ++t) {
      Frag vf;
      load_frag_b(vf, vrow0 + (size_t)(t * 16 + M) * N_ + kbase, hi);
      O[t] = wmma_bf16(pa, vf, O[t]);
    }
  }
  // finalize and store merged [B,N,C] bf16
#pragma unroll
  for (int t = 0; t < 4; ++t)
#pragma unroll
    for (int r = 0; r < 8; ++r) {
      const float v = O[t][r] / lrun[r];
      const int n = qbase + r + 8 * hi;
      const int c = h * 64 + t * 16 + M;
      Mbf[((size_t)b * N_ + n) * C_ + c] = f2bf(v);
    }
}

// ---------------- Kernel 3: out = merged @ merged^T ----------------
__global__ __launch_bounds__(128) void outer_kernel(
    const unsigned short* __restrict__ Mbf, float* __restrict__ out) {
  __shared__ unsigned short As[16][C_];
  const int blk = blockIdx.x;  // B * 64 row tiles
  const int b = blk >> 6, rt = blk & 63;
  const int tid = threadIdx.x;
  const int wave = tid >> 5, lane = tid & 31;
  const int M = lane & 15, hi = lane >> 4, Kb = hi * 8;

  const unsigned short* srcrow = Mbf + ((size_t)b * N_ + rt * 16) * C_;
#ifdef HAVE_ASYNC_LDS
  {
    // Async DMA the 16KB bf16 A-tile straight into LDS (no VGPR round-trip).
    const char* g = (const char*)srcrow;
    char* l = (char*)&As[0][0];
    for (int i = tid; i < (16 * C_ * 2) / 16; i += 128) {
      __builtin_amdgcn_global_load_async_to_lds_b128(
          (glb_v4i_p)(uintptr_t)(g + (size_t)i * 16),
          (lds_v4i_p)(uintptr_t)(l + (size_t)i * 16),
          0, 0);
    }
    __builtin_amdgcn_s_wait_asynccnt(0);
  }
#else
  {
    const unsigned int* src = (const unsigned int*)srcrow;
    unsigned int* dstl = (unsigned int*)&As[0][0];
    for (int i = tid; i < 16 * C_ / 2; i += 128) dstl[i] = src[i];
  }
#endif
  __syncthreads();

  for (int ct = wave; ct < 64; ct += 4) {
    v8f acc = {};
    for (int kk = 0; kk < 16; ++kk) {
      Frag a, bf_;
      load_frag_a(a, &As[M][kk * 32], Kb);
      load_frag_b(bf_, Mbf + ((size_t)b * N_ + ct * 16 + M) * C_ + kk * 32, hi);
      acc = wmma_bf16(a, bf_, acc);
    }
#pragma unroll
    for (int r = 0; r < 8; ++r)
      out[((size_t)b * N_ + rt * 16 + r + 8 * hi) * N_ + ct * 16 + M] = acc[r];
  }
}

extern "C" void kernel_launch(void* const* d_in, const int* in_sizes, int n_in,
                              void* d_out, int out_size, void* d_ws, size_t ws_size,
                              hipStream_t stream) {
  const float* x    = (const float*)d_in[0];
  const float* bias = (const float*)d_in[1];
  const int*   mask = (const int*)d_in[2];
  const float* Wq   = (const float*)d_in[3];
  const float* bq   = (const float*)d_in[4];
  const float* Wk   = (const float*)d_in[5];
  const float* bk   = (const float*)d_in[6];
  const float* Wv   = (const float*)d_in[7];
  const float* bv   = (const float*)d_in[8];
  float* out = (float*)d_out;

  char* ws = (char*)d_ws;
  unsigned short* Wbf  = (unsigned short*)ws;                       // 3*512*512 bf16
  unsigned short* Qbf  = (unsigned short*)(ws + (size_t)3 * C_ * C_ * 2);
  unsigned short* Kbf  = Qbf + (size_t)B_ * H_ * N_ * D_;
  unsigned short* VbfT = Kbf + (size_t)B_ * H_ * N_ * D_;
  unsigned short* Mbf  = VbfT + (size_t)B_ * H_ * N_ * D_;

  convert_w_kernel<<<(C_ * C_ + 255) / 256, 256, 0, stream>>>(Wq, Wk, Wv, Wbf);
  qkv_kernel<<<(B_ * N_) / 16, 128, 0, stream>>>(x, Wbf, bq, bk, bv, Qbf, Kbf, VbfT);
  attn_kernel<<<B_ * H_ * (N_ / 64), 128, 0, stream>>>(Qbf, Kbf, VbfT, bias, mask, Mbf);
  outer_kernel<<<B_ * (N_ / 16), 128, 0, stream>>>(Mbf, out);
}
